// GraphAttentionWithEdgeBias_50491635532402
// MI455X (gfx1250) — matrix operand order repaired and checked by reading, not compiled, not run on hardware
//
#include <hip/hip_runtime.h>
#include <hip/hip_bf16.h>

// Problem constants
#define B_   1024
#define S_   128
#define D_   256
#define E_   128
#define NH_  8
#define HD_  32

typedef __bf16 bf16;
typedef __attribute__((ext_vector_type(16))) __bf16        v16bf;
typedef __attribute__((ext_vector_type(8)))  float         v8f;
typedef __attribute__((ext_vector_type(4)))  float         v4f;
typedef __attribute__((ext_vector_type(4)))  unsigned int  v4u;

union FragU  { v16bf v; v4u q[2]; };
union PackU  { bf16 h[16]; v4u q[2]; };

static __device__ inline v8f zero_v8f() {
  v8f z;
#pragma unroll
  for (int i = 0; i < 8; ++i) z[i] = 0.0f;
  return z;
}

// ---------------------------------------------------------------------------
// Kernel 1: transpose Wk/Wv/Wo (f32 [K][N]) -> bf16 [N][K] so WMMA B-fragments
// are contiguous 32B loads per lane (contiguous k order, same as the A fragments).
// ---------------------------------------------------------------------------
__global__ __launch_bounds__(256)
void wtrans_kernel(const float* __restrict__ Wk, const float* __restrict__ Wv,
                   const float* __restrict__ Wo,
                   bf16* __restrict__ WkT, bf16* __restrict__ WvT,
                   bf16* __restrict__ WoT) {
  int id  = blockIdx.x * 256 + threadIdx.x;   // 0 .. 3*65536-1
  int mat = id >> 16;
  int r   = id & 0xFFFF;
  int n   = r >> 8, k = r & 255;
  const float* W = (mat == 0) ? Wk : ((mat == 1) ? Wv : Wo);
  bf16*        T = (mat == 0) ? WkT : ((mat == 1) ? WvT : WoT);
  T[n * 256 + k] = (bf16)W[k * 256 + n];
}

// ---------------------------------------------------------------------------
// Kernel 2: edge modulation GEMVs per batch.
//   gate=sigmoid(e@Wg+bg), scale1=1+tanh(e@Ws+bs), shift=e@Wsh+bsh,
//   ebias=e@Wb+bb, hscale=tanh(e@Wh+bh)
// ---------------------------------------------------------------------------
__global__ __launch_bounds__(256)
void edgemod_kernel(const float* __restrict__ ef,
                    const float* __restrict__ Wgate, const float* __restrict__ bgate,
                    const float* __restrict__ Wscale, const float* __restrict__ bscale,
                    const float* __restrict__ Wshift, const float* __restrict__ bshift,
                    const float* __restrict__ Wbias, const float* __restrict__ bbias,
                    const float* __restrict__ Whead, const float* __restrict__ bhead,
                    float* __restrict__ gate, float* __restrict__ scale1,
                    float* __restrict__ shift, float* __restrict__ ebias,
                    float* __restrict__ hscale) {
  __shared__ float e[E_];
  int b = blockIdx.x, t = threadIdx.x;
  if (t < E_) e[t] = ef[b * E_ + t];
  __syncthreads();

  float g = bgate[t], sc = bscale[t], sh = bshift[t];
#pragma unroll 4
  for (int k = 0; k < E_; ++k) {
    float ev = e[k];
    g  += ev * Wgate[k * D_ + t];
    sc += ev * Wscale[k * D_ + t];
    sh += ev * Wshift[k * D_ + t];
  }
  gate[b * D_ + t]   = 1.0f / (1.0f + __expf(-g));
  scale1[b * D_ + t] = 1.0f + tanhf(sc);
  shift[b * D_ + t]  = sh;

  if (t < NH_) {
    float eb = bbias[t], hs = bhead[t];
#pragma unroll 4
    for (int k = 0; k < E_; ++k) {
      float ev = e[k];
      eb += ev * Wbias[k * NH_ + t];
      hs += ev * Whead[k * NH_ + t];
    }
    ebias[b * NH_ + t]  = eb;
    hscale[b * NH_ + t] = tanhf(hs);
  }
}

// ---------------------------------------------------------------------------
// Kernel 3 (the big one): per-batch fused K/V GEMM (WMMA bf16) -> LDS,
// V modulation, 2-query-row attention, softmax, attn@V -> bf16 attn_out.
// 256 threads = 8 waves. Waves 0-3 compute K columns, waves 4-7 compute V.
// LDS: A-stage 8.4KB + K/V 128KB + xrows 2KB + qrows 2KB + attn 8KB ~ 148KB.
// ---------------------------------------------------------------------------
#define ASTRIDE 264   // 256 + 8 bf16 pad -> conflict-free ds_load_b128

__global__ __launch_bounds__(256)
void batch_attn_kernel(const float* __restrict__ x_all,      // [B][128][256]
                       const float* __restrict__ Wq, const float* __restrict__ bq,
                       const bf16* __restrict__ WkT, const bf16* __restrict__ WvT,
                       const float* __restrict__ bk, const float* __restrict__ bv,
                       const float* __restrict__ gate, const float* __restrict__ scale1,
                       const float* __restrict__ shift,
                       const float* __restrict__ ebias, const float* __restrict__ hscale,
                       bf16* __restrict__ attn_out)           // [B*2][256]
{
  __shared__ bf16  Astage[16 * ASTRIDE];
  __shared__ bf16  KV[2 * NH_ * S_ * HD_];     // [mat][h][s][hd], 128 KB
  __shared__ float xr[2 * D_];                 // x rows 63 and 127
  __shared__ float qr[2 * D_];                 // q rows
  __shared__ float attnw[NH_ * 2 * S_];        // normalized attention weights

  const int b    = blockIdx.x;
  const int tid  = threadIdx.x;
  const int lane = tid & 31;
  const int w    = tid >> 5;
  const float* x = x_all + (size_t)b * S_ * D_;

  // stage the two surviving query-source rows
  xr[tid]       = x[63 * D_ + tid];
  xr[D_ + tid]  = x[127 * D_ + tid];

  // ---- K / V GEMM phase ----------------------------------------------------
  const int   matkv = w >> 2;                  // 0 = K, 1 = V
  const int   cb    = (w & 3) * 64;            // 64-col strip within 256
  const bf16* WT    = matkv ? WvT : WkT;

  float badd[4], g1[4], sc1[4], sh1[4];
#pragma unroll
  for (int nt = 0; nt < 4; ++nt) {
    int col = cb + nt * 16 + (lane & 15);
    if (matkv == 0) {
      badd[nt] = bk[col]; g1[nt] = 1.0f; sc1[nt] = 1.0f; sh1[nt] = 0.0f;
    } else {
      badd[nt] = bv[col];
      g1[nt]  = gate[b * D_ + col];
      sc1[nt] = scale1[b * D_ + col];
      sh1[nt] = shift[b * D_ + col];
    }
  }

  for (int mt = 0; mt < 8; ++mt) {
    __syncthreads();                           // protect Astage reuse
    // cooperative stage: 16 rows x 256 f32 -> bf16 LDS (16 elems / thread)
    {
      int row = tid >> 4, kb = (tid & 15) * 16;
      const v4f* src = (const v4f*)(x + (size_t)(mt * 16 + row) * D_ + kb);
      // prefetch the same lane's chunk of the NEXT tile into cache
      if (mt < 7)
        __builtin_prefetch(x + (size_t)((mt + 1) * 16 + row) * D_ + kb, 0, 3);
      PackU pk;
#pragma unroll
      for (int i = 0; i < 4; ++i) {
        v4f f = src[i];
        pk.h[4 * i + 0] = (bf16)f.x; pk.h[4 * i + 1] = (bf16)f.y;
        pk.h[4 * i + 2] = (bf16)f.z; pk.h[4 * i + 3] = (bf16)f.w;
      }
      *(v4u*)&Astage[row * ASTRIDE + kb]     = pk.q[0];
      *(v4u*)&Astage[row * ASTRIDE + kb + 8] = pk.q[1];
    }
    __syncthreads();

    v8f acc[4];
#pragma unroll
    for (int nt = 0; nt < 4; ++nt) acc[nt] = zero_v8f();

    const int M  = lane & 15;
    const int kh = (lane < 16) ? 0 : 16;       // A K-half (contiguous, matches B)
    const int khB = (lane < 16) ? 0 : 16;      // B K-half (contiguous, matches A)
#pragma unroll
    for (int kk = 0; kk < 8; ++kk) {           // K loop, 32 per step
      FragU a;
      a.q[0] = *(const v4u*)&Astage[M * ASTRIDE + kk * 32 + kh];
      a.q[1] = *(const v4u*)&Astage[M * ASTRIDE + kk * 32 + 8 + kh];
#pragma unroll
      for (int nt = 0; nt < 4; ++nt) {
        int n = cb + nt * 16 + (lane & 15);
        FragU bf;
        const v4u* bp = (const v4u*)&WT[n * D_ + kk * 32 + khB];
        bf.q[0] = bp[0];
        bf.q[1] = bp[1];
        acc[nt] = __builtin_amdgcn_wmma_f32_16x16x32_bf16(
            false, a.v, false, bf.v, (short)0, acc[nt], false, false);
      }
    }

    // epilogue: bias (+ V modulation) -> KV LDS as [mat][h][s][hd] bf16
#pragma unroll
    for (int nt = 0; nt < 4; ++nt) {
      int col = cb + nt * 16 + (lane & 15);
      int h = col >> 5, hd = col & 31;
      bf16* dst = &KV[(matkv * NH_ + h) * (S_ * HD_) + hd];
#pragma unroll
      for (int vr = 0; vr < 8; ++vr) {
        int srow = mt * 16 + vr + 8 * (lane >> 4);
        float v = acc[nt][vr] + badd[nt];
        if (matkv) v = (v * sc1[nt] + sh1[nt]) * g1[nt];
        dst[srow * HD_] = (bf16)v;
      }
    }
  }
  __syncthreads();

  // ---- Q phase: only rows 63 and 127 are needed ---------------------------
  {
    float a0 = bq[tid], a1 = a0;
#pragma unroll 4
    for (int k = 0; k < D_; ++k) {
      float wv = Wq[k * D_ + tid];
      a0 += xr[k] * wv;
      a1 += xr[D_ + k] * wv;
    }
    qr[tid]      = a0;
    qr[D_ + tid] = a1;
  }
  __syncthreads();

  // ---- Attention: wave w == head h ----------------------------------------
  {
    const int   h    = w;
    const float hs   = hscale[b * NH_ + h];
    const float eb   = ebias[b * NH_ + h];
    const float sfac = (1.0f + hs) * 0.17677669529663687f;  // (1+hs)/sqrt(32)
    const bf16* Kh = &KV[(0 * NH_ + h) * (S_ * HD_)];

    float s0[4], s1[4];
#pragma unroll
    for (int c = 0; c < 4; ++c) {
      int kp = c * 32 + lane;
      const bf16* krow = &Kh[kp * HD_];
      float d0 = 0.f, d1 = 0.f;
#pragma unroll
      for (int d = 0; d < HD_; ++d) {
        float kv = (float)krow[d];
        d0 += qr[h * HD_ + d] * kv;
        d1 += qr[D_ + h * HD_ + d] * kv;
      }
      // sparse edge bias: query row 63 biases key 127; row 127 biases key 63
      s0[c] = d0 * sfac + ((kp == 127) ? eb : 0.0f);
      s1[c] = d1 * sfac + ((kp == 63) ? eb : 0.0f);
    }

    // softmax over 128 keys (distributed 4 per lane), wave32 reductions
    float m0 = fmaxf(fmaxf(s0[0], s0[1]), fmaxf(s0[2], s0[3]));
    float m1 = fmaxf(fmaxf(s1[0], s1[1]), fmaxf(s1[2], s1[3]));
#pragma unroll
    for (int off = 16; off > 0; off >>= 1) {
      m0 = fmaxf(m0, __shfl_xor(m0, off, 32));
      m1 = fmaxf(m1, __shfl_xor(m1, off, 32));
    }
    float p0[4], p1[4], sum0 = 0.f, sum1 = 0.f;
#pragma unroll
    for (int c = 0; c < 4; ++c) {
      p0[c] = __expf(s0[c] - m0); sum0 += p0[c];
      p1[c] = __expf(s1[c] - m1); sum1 += p1[c];
    }
#pragma unroll
    for (int off = 16; off > 0; off >>= 1) {
      sum0 += __shfl_xor(sum0, off, 32);
      sum1 += __shfl_xor(sum1, off, 32);
    }
    float inv0 = 1.0f / sum0, inv1 = 1.0f / sum1;
#pragma unroll
    for (int c = 0; c < 4; ++c) {
      int kp = c * 32 + lane;
      attnw[h * 2 * S_ + kp]      = p0[c] * inv0;
      attnw[h * 2 * S_ + S_ + kp] = p1[c] * inv1;
    }
  }
  __syncthreads();

  // ---- attn @ V: lane = hd ------------------------------------------------
  {
    const int h = w, d = lane;
    const bf16* Vh = &KV[(NH_ + h) * (S_ * HD_)];
    float o0 = 0.f, o1 = 0.f;
#pragma unroll 4
    for (int kp = 0; kp < S_; ++kp) {
      float vv = (float)Vh[kp * HD_ + d];
      o0 += attnw[h * 2 * S_ + kp] * vv;
      o1 += attnw[h * 2 * S_ + S_ + kp] * vv;
    }
    attn_out[(size_t)(b * 2 + 0) * D_ + h * HD_ + d] = (bf16)o0;
    attn_out[(size_t)(b * 2 + 1) * D_ + h * HD_ + d] = (bf16)o1;
  }
}

// ---------------------------------------------------------------------------
// Kernel 4: final WMMA GEMM: (2048x256 bf16) @ WoT + bo -> f32 d_out.
// 128 blocks, each one 16-row M-tile; 8 waves x 32-col strips.
// ---------------------------------------------------------------------------
__global__ __launch_bounds__(256)
void final_gemm_kernel(const bf16* __restrict__ Ain,   // [2048][256] bf16
                       const bf16* __restrict__ WoT,   // [256][256] bf16 (transposed)
                       const float* __restrict__ bo,
                       float* __restrict__ out)        // [2048][256] f32
{
  __shared__ bf16 Astage[16 * ASTRIDE];
  const int tid = threadIdx.x, lane = tid & 31, w = tid >> 5;
  const int cb = w * 32;
  const int row0 = blockIdx.x * 16;

  float bo2[2];
#pragma unroll
  for (int nt = 0; nt < 2; ++nt) bo2[nt] = bo[cb + nt * 16 + (lane & 15)];

  const int M   = lane & 15;
  const int kh  = (lane < 16) ? 0 : 16;
  const int khB = (lane < 16) ? 0 : 16;

  // stage the 16-row bf16 tile
  {
    int r = tid >> 4, kb = (tid & 15) * 16;
    const v4u* src = (const v4u*)&Ain[(size_t)(row0 + r) * D_ + kb];
    v4u u0 = src[0], u1 = src[1];
    *(v4u*)&Astage[r * ASTRIDE + kb]     = u0;
    *(v4u*)&Astage[r * ASTRIDE + kb + 8] = u1;
  }
  __syncthreads();

  v8f acc[2];
  acc[0] = zero_v8f(); acc[1] = zero_v8f();
#pragma unroll
  for (int kk = 0; kk < 8; ++kk) {
    FragU a;
    a.q[0] = *(const v4u*)&Astage[M * ASTRIDE + kk * 32 + kh];
    a.q[1] = *(const v4u*)&Astage[M * ASTRIDE + kk * 32 + 8 + kh];
#pragma unroll
    for (int nt = 0; nt < 2; ++nt) {
      int n = cb + nt * 16 + (lane & 15);
      FragU bf;
      const v4u* bp = (const v4u*)&WoT[n * D_ + kk * 32 + khB];
      bf.q[0] = bp[0];
      bf.q[1] = bp[1];
      acc[nt] = __builtin_amdgcn_wmma_f32_16x16x32_bf16(
          false, a.v, false, bf.v, (short)0, acc[nt], false, false);
    }
  }
#pragma unroll
  for (int nt = 0; nt < 2; ++nt) {
    int col = cb + nt * 16 + (lane & 15);
#pragma unroll
    for (int vr = 0; vr < 8; ++vr) {
      int rr = row0 + vr + 8 * (lane >> 4);
      out[(size_t)rr * D_ + col] = acc[nt][vr] + bo2[nt];
    }
  }
}

// ---------------------------------------------------------------------------
extern "C" void kernel_launch(void* const* d_in, const int* in_sizes, int n_in,
                              void* d_out, int out_size, void* d_ws, size_t ws_size,
                              hipStream_t stream) {
  const float* node_features = (const float*)d_in[0];
  const float* edge_features = (const float*)d_in[1];
  const float* Wq = (const float*)d_in[2];  const float* bq = (const float*)d_in[3];
  const float* Wk = (const float*)d_in[4];  const float* bk = (const float*)d_in[5];
  const float* Wv = (const float*)d_in[6];  const float* bv = (const float*)d_in[7];
  const float* Wo = (const float*)d_in[8];  const float* bo = (const float*)d_in[9];
  const float* Wbias = (const float*)d_in[10]; const float* bbias = (const float*)d_in[11];
  const float* Wgate = (const float*)d_in[12]; const float* bgate = (const float*)d_in[13];
  const float* Wscale = (const float*)d_in[14]; const float* bscale = (const float*)d_in[15];
  const float* Wshift = (const float*)d_in[16]; const float* bshift = (const float*)d_in[17];
  const float* Whead = (const float*)d_in[18]; const float* bhead = (const float*)d_in[19];
  float* out = (float*)d_out;

  // workspace carve-up (all offsets 256B-aligned)
  uintptr_t p = (uintptr_t)d_ws;
  bf16* WkT = (bf16*)p;        p += (size_t)D_ * D_ * 2;
  bf16* WvT = (bf16*)p;        p += (size_t)D_ * D_ * 2;
  bf16* WoT = (bf16*)p;        p += (size_t)D_ * D_ * 2;
  float* gate   = (float*)p;   p += (size_t)B_ * D_ * 4;
  float* scale1 = (float*)p;   p += (size_t)B_ * D_ * 4;
  float* shift  = (float*)p;   p += (size_t)B_ * D_ * 4;
  float* ebias  = (float*)p;   p += (size_t)B_ * NH_ * 4;
  float* hscale = (float*)p;   p += (size_t)B_ * NH_ * 4;
  bf16* attn_out = (bf16*)p;   p += (size_t)B_ * 2 * D_ * 2;

  wtrans_kernel<<<768, 256, 0, stream>>>(Wk, Wv, Wo, WkT, WvT, WoT);

  edgemod_kernel<<<B_, 256, 0, stream>>>(edge_features,
      Wgate, bgate, Wscale, bscale, Wshift, bshift,
      Wbias, bbias, Whead, bhead,
      gate, scale1, shift, ebias, hscale);

  batch_attn_kernel<<<B_, 256, 0, stream>>>(node_features,
      Wq, bq, WkT, WvT, bk, bv,
      gate, scale1, shift, ebias, hscale, attn_out);

  final_gemm_kernel<<<128, 256, 0, stream>>>(attn_out, WoT, bo, out);
}
